// GeneralizedCausalAttention_1073741824369
// MI455X (gfx1250) — matrix operand order repaired and checked by reading, not compiled, not run on hardware
//
#include <hip/hip_runtime.h>
#include <hip/hip_bf16.h>

typedef _Float16 f16;
typedef __attribute__((ext_vector_type(16))) _Float16 v16h;
typedef __attribute__((ext_vector_type(8)))  _Float16 v8h;
typedef __attribute__((ext_vector_type(8)))  float    v8f;
typedef unsigned int u32x4 __attribute__((ext_vector_type(4)));
typedef int i32x8 __attribute__((ext_vector_type(8)));
typedef int i32x4 __attribute__((ext_vector_type(4)));

#define DIMC 1024
#define SEQL 2048
#define NBATCH 4
#define NHEADS 16
#define HD 64
#define MTOT (NBATCH * SEQL)   // 8192

#if defined(__has_builtin)
#if __has_builtin(__builtin_amdgcn_tensor_load_to_lds)
#define HAVE_TDM 1
#endif
#endif

// ---------------------------------------------------------------------------
// WMMA wrapper: D = A(16x32 f16) * B(32x16 f16) + C(16x16 f32)
// ---------------------------------------------------------------------------
__device__ __forceinline__ v8f wmma16(v16h a, v16h b, v8f c) {
  return __builtin_amdgcn_wmma_f32_16x16x32_f16(
      /*neg_a=*/false, a, /*neg_b=*/false, b,
      /*c_mod=*/(short)0, c, /*reuse_a=*/false, /*reuse_b=*/false);
}

// A fragment (16x32 f16) from row-major LDS [row][lda], K window [koff,koff+32).
// ISA: lanes 0-15 hold K={0..7,16..23}, lanes 16-31 hold K={8..15,24..31}.
__device__ __forceinline__ v16h load_frag_a(const f16* lds, int rbase, int lda,
                                            int koff, int lane) {
  int r  = rbase + (lane & 15);
  int hi = (lane >> 4) & 1;
  const f16* p = lds + r * lda + koff;
  v8h lo = *(const v8h*)(p + hi * 8);
  v8h hh = *(const v8h*)(p + 16 + hi * 8);
  return __builtin_shufflevector(lo, hh, 0,1,2,3,4,5,6,7,8,9,10,11,12,13,14,15);
}

// B fragment (32x16 f16) from LDS [col][ldc] ("K contiguous per column").
// ISA: lanes 0-15 hold K={0..7,16..23}, lanes 16-31 hold K={8..15,24..31}.
__device__ __forceinline__ v16h load_frag_b(const f16* lds, int cbase, int ldc,
                                            int koff, int lane) {
  int c  = cbase + (lane & 15);
  int hi = (lane >> 4) & 1;
  const f16* p = lds + c * ldc + koff;
  v8h lo = *(const v8h*)(p + hi * 8);
  v8h hh = *(const v8h*)(p + 16 + hi * 8);
  return __builtin_shufflevector(lo, hh, 0,1,2,3,4,5,6,7,8,9,10,11,12,13,14,15);
}

// ---------------------------------------------------------------------------
// CDNA5 async memory->LDS copy (16B per active lane), ASYNCcnt tracked.
// LDS generic pointers carry the LDS offset in addr[31:0] (ISA 10.2), so a
// 32-bit truncation yields the DS address for the async op.
// ---------------------------------------------------------------------------
__device__ __forceinline__ void async_copy16(void* lds, const void* g) {
  unsigned l = (unsigned)(unsigned long long)(uintptr_t)lds;
  unsigned long long ga = (unsigned long long)(uintptr_t)g;
  asm volatile("global_load_async_to_lds_b128 %0, %1, off"
               :: "v"(l), "v"(ga) : "memory");
}
__device__ __forceinline__ void wait_async0() {
  asm volatile("s_wait_asynccnt 0x0" ::: "memory");
}

// ---------------------------------------------------------------------------
// Tensor Data Mover: 1-D contiguous copy of nelem f16 elements, global->LDS.
// D# per ISA §8.3/8.4: group0 = {count=1 | lds_addr | global_addr | type=2},
// group1 = {data_size=2B, tensor_dim0=nelem, tensor_dim1=1, tile_dim0=nelem,
//           tile_dim1/2=0 (unused), tensor_dim0_stride=nelem}.
// Groups 2/3 unused for a <=2-D tensor -> zero-filled (6-arg builtin).
// Issued once per workgroup (wave 0); EXEC is ignored by TENSOR_LOAD_TO_LDS.
// ---------------------------------------------------------------------------
#ifdef HAVE_TDM
__device__ __forceinline__ void tdm_copy_1d(void* lds, const void* gsrc,
                                            unsigned nelem) {
  unsigned long long ga = (unsigned long long)(uintptr_t)gsrc;
  unsigned la = (unsigned)(uintptr_t)lds;
  u32x4 g0;
  g0[0] = 1u;                                          // count=1, user mode
  g0[1] = la;                                          // lds_addr (bytes)
  g0[2] = (unsigned)(ga & 0xffffffffu);                // global_addr[31:0]
  g0[3] = (unsigned)((ga >> 32) & 0x1ffffffu) | (2u << 30);  // addr[56:32]|type=2
  i32x8 g1;
  g1[0] = 0x00010000;                                  // data_size=1 -> 2 bytes
  g1[1] = (int)((nelem & 0xffffu) << 16);              // tensor_dim0[15:0]
  g1[2] = (int)(((nelem >> 16) & 0xffffu) | (1u << 16)); // dim0[31:16]|dim1=1
  g1[3] = (int)((nelem & 0xffffu) << 16);              // tile_dim0
  g1[4] = 0;                                           // tile_dim1/2 unused
  g1[5] = (int)nelem;                                  // tensor_dim0_stride lo
  g1[6] = 0;
  g1[7] = 0;
  i32x4 z4 = {0, 0, 0, 0};
  i32x8 z8 = {0, 0, 0, 0, 0, 0, 0, 0};
  __builtin_amdgcn_tensor_load_to_lds(g0, g1, z4, z4, z8, 0);
}
#endif

// ---------------------------------------------------------------------------
// Kernel 0: f32 -> f16 cast
// ---------------------------------------------------------------------------
__global__ void cast_f32_to_f16(const float* __restrict__ in,
                                f16* __restrict__ out, int n) {
  int i = blockIdx.x * blockDim.x + threadIdx.x;
  int stride = gridDim.x * blockDim.x;
  for (; i < n; i += stride) out[i] = (f16)in[i];
}

// ---------------------------------------------------------------------------
// Kernel 1: QKV GEMM + fused per-head LayerNorm (q pre-scaled by hd^-0.5).
// 64x64 tile, K staged through double-buffered LDS via async-to-LDS copies.
// LN stats reduced 4-way (4 threads/row, 16-col partials, one LDS combine).
// ---------------------------------------------------------------------------
__global__ __launch_bounds__(256) void qkv_ln_kernel(
    const f16* __restrict__ xh, const f16* __restrict__ wh,
    const float* __restrict__ qg, const float* __restrict__ qb,
    const float* __restrict__ kg, const float* __restrict__ kb,
    f16* __restrict__ Qw, f16* __restrict__ Kw, f16* __restrict__ Vw) {
  __shared__ __align__(16) f16  As[2][64 * 32];
  __shared__ __align__(16) f16  Bs[2][64 * 32];
  __shared__ __align__(16) float Cs[64 * 64];
  __shared__ float red1[256], red2[256];

  const int m0 = blockIdx.x * 64;
  const int e0 = blockIdx.y * 64;
  const int tid = threadIdx.x, lane = tid & 31, wave = tid >> 5;
  const int mi = wave & 3;
  const int ni0 = (wave >> 2) * 2;

  v8f acc0 = {}, acc1 = {};
  const int lr = tid >> 2;          // 0..63
  const int lk = (tid & 3) * 8;     // 0,8,16,24

  const f16* arow = xh + (size_t)(m0 + lr) * DIMC + lk;
  const f16* brow = wh + (size_t)(e0 + lr) * DIMC + lk;

  // prologue: fill buffer 0
  async_copy16(&As[0][lr * 32 + lk], arow);
  async_copy16(&Bs[0][lr * 32 + lk], brow);
  wait_async0();
  __syncthreads();

  int cur = 0;
  for (int k0 = 0; k0 < DIMC; k0 += 32) {
    int nxt = cur ^ 1;
    if (k0 + 32 < DIMC) {
      async_copy16(&As[nxt][lr * 32 + lk], arow + k0 + 32);
      async_copy16(&Bs[nxt][lr * 32 + lk], brow + k0 + 32);
      __builtin_prefetch(arow + k0 + 64, 0, 0);
    }
    v16h a  = load_frag_a(As[cur], mi * 16, 32, 0, lane);
    v16h b0 = load_frag_b(Bs[cur], ni0 * 16, 32, 0, lane);
    v16h b1 = load_frag_b(Bs[cur], ni0 * 16 + 16, 32, 0, lane);
    acc0 = wmma16(a, b0, acc0);
    acc1 = wmma16(a, b1, acc1);
    wait_async0();
    __syncthreads();
    cur = nxt;
  }

  const int colA = lane & 15, hiL = lane >> 4;
  for (int r = 0; r < 8; ++r) {
    int row = mi * 16 + r + 8 * hiL;
    Cs[row * 64 + ni0 * 16 + colA]      = acc0[r];
    Cs[row * 64 + ni0 * 16 + 16 + colA] = acc1[r];
  }
  __syncthreads();

  const int comp = e0 >> 10;           // 0=q 1=k 2=v
  const int head = (e0 & 1023) >> 6;
  const int n    = m0 >> 11;
  const int l0   = m0 & (SEQL - 1);

  // 4 threads per row: 16-column partial sums / sums-of-squares
  const int row = tid >> 2;
  const int cb  = (tid & 3) * 16;
  float s = 0.f, s2 = 0.f;
  for (int c = 0; c < 16; ++c) {
    float v = Cs[row * 64 + cb + c];
    s += v;
    s2 += v * v;
  }
  red1[tid] = s;
  red2[tid] = s2;
  __syncthreads();
  float ssum  = red1[row * 4 + 0] + red1[row * 4 + 1] +
                red1[row * 4 + 2] + red1[row * 4 + 3];
  float s2sum = red2[row * 4 + 0] + red2[row * 4 + 1] +
                red2[row * 4 + 2] + red2[row * 4 + 3];
  float mu = ssum * (1.f / 64.f);
  float ri = rsqrtf(fmaxf(s2sum * (1.f / 64.f) - mu * mu, 0.f) + 1e-5f);

  const size_t base = ((size_t)(n * NHEADS + head) * SEQL + (l0 + row)) * HD;
  const float qscale = 0.125f;  // hd^-0.5
  if (comp == 0) {
    for (int c = 0; c < 16; ++c) {
      int d = cb + c;
      float v = (Cs[row * 64 + d] - mu) * ri * qg[d] + qb[d];
      Qw[base + d] = (f16)(v * qscale);
    }
  } else if (comp == 1) {
    for (int c = 0; c < 16; ++c) {
      int d = cb + c;
      float v = (Cs[row * 64 + d] - mu) * ri * kg[d] + kb[d];
      Kw[base + d] = (f16)v;
    }
  } else {
    for (int c = 0; c < 16; ++c) {
      int d = cb + c;
      Vw[base + d] = (f16)Cs[row * 64 + d];
    }
  }
}

// ---------------------------------------------------------------------------
// Kernel 2: flash attention.  Q and K tiles are contiguous 8KB blocks ->
// moved by the Tensor Data Mover (wave 0 issues, TENSORcnt wait + barrier).
// V is staged transposed ([d][kj]) so it loads as a legal B fragment.
// Online softmax reduced 4-way per row (4 threads x 16 columns).
// ---------------------------------------------------------------------------
__global__ __launch_bounds__(256) void attn_kernel(
    const f16* __restrict__ Qw, const f16* __restrict__ Kw,
    const f16* __restrict__ Vw, f16* __restrict__ Ow) {
  __shared__ __align__(16) f16  Qs[64 * 64];
  __shared__ __align__(16) f16  Ks[64 * 64];   // [kj][d]
  __shared__ __align__(16) f16  Vs[64 * 64];   // transposed: [d][kj]
  __shared__ __align__(16) float Ss[64 * 64];
  __shared__ __align__(16) f16  Ps[64 * 64];   // probs, [qi][kj]
  __shared__ float mrow[64], lrow[64], arow_s[64];
  __shared__ float red[256];

  const int qb = blockIdx.x * 64;
  const int nh = blockIdx.y;
  const f16* Qp = Qw + ((size_t)nh * SEQL + qb) * HD;
  const f16* Kp = Kw + (size_t)nh * SEQL * HD;
  const f16* Vp = Vw + (size_t)nh * SEQL * HD;

  const int tid = threadIdx.x, lane = tid & 31, wave = tid >> 5;
  const int mi = wave & 3, ni0 = (wave >> 2) * 2;
  const int colA = lane & 15, hiL = lane >> 4;
  const int lr = tid >> 2;                   // 0..63
  const int lk = (tid & 3) * 16;             // 0,16,32,48
  const int srow = tid >> 2;                 // softmax row (0..63)
  const int scb  = (tid & 3) * 16;           // softmax column base

#ifdef HAVE_TDM
  if (wave == 0) {
    tdm_copy_1d(Qs, Qp, 64 * 64);
    __builtin_amdgcn_s_wait_tensorcnt(0);
  }
#else
  *(u32x4*)(Qs + lr * 64 + lk)     = *(const u32x4*)(Qp + lr * 64 + lk);
  *(u32x4*)(Qs + lr * 64 + lk + 8) = *(const u32x4*)(Qp + lr * 64 + lk + 8);
#endif
  if (tid < 64) { mrow[tid] = -1e30f; lrow[tid] = 0.f; }

  v8f o0 = {}, o1 = {};

  for (int kb = 0; kb <= qb; kb += 64) {
    __syncthreads();   // previous-tile LDS reads done; also covers Qs/mrow init
#ifdef HAVE_TDM
    if (wave == 0) tdm_copy_1d(Ks, Kp + (size_t)kb * 64, 64 * 64);
#else
    *(u32x4*)(Ks + lr * 64 + lk)     = *(const u32x4*)(Kp + (size_t)(kb + lr) * 64 + lk);
    *(u32x4*)(Ks + lr * 64 + lk + 8) = *(const u32x4*)(Kp + (size_t)(kb + lr) * 64 + lk + 8);
#endif
    // V tile, transposed into [d][kj]
    {
      v8h t0 = *(const v8h*)(Vp + (size_t)(kb + lr) * 64 + lk);
      v8h t1 = *(const v8h*)(Vp + (size_t)(kb + lr) * 64 + lk + 8);
      for (int c = 0; c < 8; ++c) Vs[(lk + c) * 64 + lr]     = t0[c];
      for (int c = 0; c < 8; ++c) Vs[(lk + 8 + c) * 64 + lr] = t1[c];
    }
    if (kb + 64 <= qb)
      __builtin_prefetch(Kp + (size_t)(kb + 64 + lr) * 64 + lk, 0, 0);
#ifdef HAVE_TDM
    if (wave == 0) __builtin_amdgcn_s_wait_tensorcnt(0);
#endif
    __syncthreads();

    // S = Q * K^T  (each wave: 2 subtiles of 16x16, d = 64 -> 2 WMMA each)
    v8f s0 = {}, s1 = {};
    for (int dk = 0; dk < 64; dk += 32) {
      v16h a  = load_frag_a(Qs, mi * 16, 64, dk, lane);
      v16h b0 = load_frag_b(Ks, ni0 * 16, 64, dk, lane);
      v16h b1 = load_frag_b(Ks, ni0 * 16 + 16, 64, dk, lane);
      s0 = wmma16(a, b0, s0);
      s1 = wmma16(a, b1, s1);
    }
    const bool diag = (kb == qb);
    for (int r = 0; r < 8; ++r) {
      int qi = mi * 16 + r + 8 * hiL;
      int c0 = ni0 * 16 + colA, c1 = c0 + 16;
      float v0 = s0[r], v1 = s1[r];
      if (diag) {
        if (c0 > qi) v0 = -1e30f;
        if (c1 > qi) v1 = -1e30f;
      }
      Ss[qi * 64 + c0] = v0;
      Ss[qi * 64 + c1] = v1;
    }
    __syncthreads();

    // online softmax, 4 threads per row --------------------------------
    // phase 1: partial max over this thread's 16 columns
    {
      float pm = -1e30f;
      for (int c = 0; c < 16; ++c) pm = fmaxf(pm, Ss[srow * 64 + scb + c]);
      red[srow * 4 + (tid & 3)] = pm;
    }
    __syncthreads();
    // phase 2: combine maxes (all 4 threads of a row are in one wave, so the
    // partial-max reads below are ordered before the partial-sum overwrite),
    // exponentiate own 16 columns, write f16 probs, store partial sums.
    {
      float mo = mrow[srow];
      float mt = fmaxf(fmaxf(red[srow * 4 + 0], red[srow * 4 + 1]),
                       fmaxf(red[srow * 4 + 2], red[srow * 4 + 3]));
      mt = fmaxf(mo, mt);
      float sum = 0.f;
      for (int c = 0; c < 16; ++c) {
        float p = __expf(Ss[srow * 64 + scb + c] - mt);
        sum += p;
        Ps[srow * 64 + scb + c] = (f16)p;
      }
      red[srow * 4 + (tid & 3)] = sum;
      if ((tid & 3) == 0) {
        mrow[srow]   = mt;
        arow_s[srow] = __expf(mo - mt);
      }
    }
    __syncthreads();
    // phase 3: one thread per row folds partial sums into l
    if ((tid & 3) == 0) {
      float sum = red[srow * 4 + 0] + red[srow * 4 + 1] +
                  red[srow * 4 + 2] + red[srow * 4 + 3];
      lrow[srow] = lrow[srow] * arow_s[srow] + sum;
    }
    __syncthreads();

    for (int r = 0; r < 8; ++r) {
      float al = arow_s[mi * 16 + r + 8 * hiL];
      o0[r] *= al;
      o1[r] *= al;
    }
    // O += P * V   (keys = 64 -> 2 WMMA per subtile)
    for (int kk = 0; kk < 64; kk += 32) {
      v16h a  = load_frag_a(Ps, mi * 16, 64, kk, lane);
      v16h b0 = load_frag_b(Vs, ni0 * 16, 64, kk, lane);
      v16h b1 = load_frag_b(Vs, ni0 * 16 + 16, 64, kk, lane);
      o0 = wmma16(a, b0, o0);
      o1 = wmma16(a, b1, o1);
    }
  }

  const int n = nh >> 4, head = nh & 15;
  for (int r = 0; r < 8; ++r) {
    int qi = mi * 16 + r + 8 * hiL;
    float linv = 1.f / lrow[qi];
    size_t orow = ((size_t)(n * SEQL + qb + qi)) * DIMC + head * HD;
    int d0 = ni0 * 16 + colA, d1 = d0 + 16;
    Ow[orow + d0] = (f16)(o0[r] * linv);
    Ow[orow + d1] = (f16)(o1[r] * linv);
  }
}

// ---------------------------------------------------------------------------
// Kernel 3: output projection  out = Oh[8192x1024] * w_proj^T + b -> f32,
// double-buffered async-to-LDS staging.
// ---------------------------------------------------------------------------
__global__ __launch_bounds__(256) void proj_kernel(
    const f16* __restrict__ Ah, const f16* __restrict__ Wh,
    const float* __restrict__ bias, float* __restrict__ out) {
  __shared__ __align__(16) f16 As[2][64 * 32];
  __shared__ __align__(16) f16 Bs[2][64 * 32];

  const int m0 = blockIdx.x * 64;
  const int e0 = blockIdx.y * 64;
  const int tid = threadIdx.x, lane = tid & 31, wave = tid >> 5;
  const int mi = wave & 3, ni0 = (wave >> 2) * 2;
  const int lr = tid >> 2, lk = (tid & 3) * 8;

  const f16* arow = Ah + (size_t)(m0 + lr) * DIMC + lk;
  const f16* brow = Wh + (size_t)(e0 + lr) * DIMC + lk;

  async_copy16(&As[0][lr * 32 + lk], arow);
  async_copy16(&Bs[0][lr * 32 + lk], brow);
  wait_async0();
  __syncthreads();

  v8f acc0 = {}, acc1 = {};
  int cur = 0;
  for (int k0 = 0; k0 < DIMC; k0 += 32) {
    int nxt = cur ^ 1;
    if (k0 + 32 < DIMC) {
      async_copy16(&As[nxt][lr * 32 + lk], arow + k0 + 32);
      async_copy16(&Bs[nxt][lr * 32 + lk], brow + k0 + 32);
    }
    v16h a  = load_frag_a(As[cur], mi * 16, 32, 0, lane);
    v16h b0 = load_frag_b(Bs[cur], ni0 * 16, 32, 0, lane);
    v16h b1 = load_frag_b(Bs[cur], ni0 * 16 + 16, 32, 0, lane);
    acc0 = wmma16(a, b0, acc0);
    acc1 = wmma16(a, b1, acc1);
    wait_async0();
    __syncthreads();
    cur = nxt;
  }

  const int colA = lane & 15, hiL = lane >> 4;
  for (int r = 0; r < 8; ++r) {
    int row = m0 + mi * 16 + r + 8 * hiL;
    int c0 = e0 + ni0 * 16 + colA, c1 = c0 + 16;
    out[(size_t)row * DIMC + c0] = acc0[r] + bias[c0];
    out[(size_t)row * DIMC + c1] = acc1[r] + bias[c1];
  }
}

// ---------------------------------------------------------------------------
// Host entry
// ---------------------------------------------------------------------------
extern "C" void kernel_launch(void* const* d_in, const int* in_sizes, int n_in,
                              void* d_out, int out_size, void* d_ws, size_t ws_size,
                              hipStream_t stream) {
  (void)in_sizes; (void)n_in; (void)out_size; (void)ws_size;
  const float* x      = (const float*)d_in[0];
  // d_in[1] = attn_mask (causal, recomputed analytically)
  const float* w_qkv  = (const float*)d_in[2];
  const float* w_proj = (const float*)d_in[3];
  const float* b_proj = (const float*)d_in[4];
  const float* q_g    = (const float*)d_in[5];
  const float* q_b    = (const float*)d_in[6];
  const float* k_g    = (const float*)d_in[7];
  const float* k_b    = (const float*)d_in[8];
  float* out = (float*)d_out;

  char* ws = (char*)d_ws;
  f16* xh     = (f16*)(ws);                    // 16 MB  (8192*1024)
  f16* wqkvh  = (f16*)(ws + (16u << 20));      //  6 MB  (3072*1024)
  f16* wprojh = (f16*)(ws + (22u << 20));      //  2 MB  (1024*1024)
  f16* Qw     = (f16*)(ws + (24u << 20));      // 16 MB  [N,H,L,64]
  f16* Kw     = (f16*)(ws + (40u << 20));      // 16 MB
  f16* Vw     = (f16*)(ws + (56u << 20));      // 16 MB
  f16* Ow     = (f16*)(ws + (72u << 20));      // 16 MB  [N,L,D]

  cast_f32_to_f16<<<2048, 256, 0, stream>>>(x, xh, MTOT * DIMC);
  cast_f32_to_f16<<<1024, 256, 0, stream>>>(w_qkv, wqkvh, 3 * DIMC * DIMC);
  cast_f32_to_f16<<<512, 256, 0, stream>>>(w_proj, wprojh, DIMC * DIMC);

  qkv_ln_kernel<<<dim3(MTOT / 64, (3 * DIMC) / 64), 256, 0, stream>>>(
      xh, wqkvh, q_g, q_b, k_g, k_b, Qw, Kw, Vw);

  attn_kernel<<<dim3(SEQL / 64, NBATCH * NHEADS), 256, 0, stream>>>(
      Qw, Kw, Vw, Ow);

  proj_kernel<<<dim3(MTOT / 64, DIMC / 64), 256, 0, stream>>>(
      Ow, wprojh, b_proj, out);
}
